// RigidNet_83786222011179
// MI455X (gfx1250) — matrix operand order repaired and checked by reading, not compiled, not run on hardware
//
#include <hip/hip_runtime.h>

// ---------------- CDNA5 WMMA plumbing (gfx1250, wave32) ----------------
typedef __attribute__((ext_vector_type(16))) _Float16 v16h;
typedef __attribute__((ext_vector_type(8)))  _Float16 h8v;
typedef __attribute__((ext_vector_type(8)))  float    v8f;
typedef __attribute__((ext_vector_type(4)))  float    f4v;

#define DEVI __device__ __forceinline__

DEVI int imin(int a, int b){ return a < b ? a : b; }

DEVI v8f wmma16(v16h a, v16h b, v8f c){
  // D = A(16x32 f16) * B(32x16 f16) + C(16x16 f32)
  return __builtin_amdgcn_wmma_f32_16x16x32_f16(false, a, false, b, (short)0, c, false, false);
}

// 16-bit A-matrix 16x32 K-index for vector element e (ISA 7.12.2)
DEVI int akk(int e, int lane){ return (e & 7) + ((e >> 3) << 4) + ((lane >> 4) << 3); }

// B fragment from packed weights: layout (((ks*nct + ct)*32 + lane)*16 + e), one 32B load.
DEVI v16h load_bp(const _Float16* __restrict__ Wp, int nct, int ks, int ct, int lane){
  return *(const v16h*)(Wp + ((((size_t)ks * nct + ct) * 32 + lane) << 4));
}

// A fragment from f16 row (global or LDS): rowk points at K=0 of this 32-wide k-chunk.
DEVI v16h a_from_f16p(const _Float16* rowk, int lane){
  const _Float16* b0 = rowk + ((lane >> 4) << 3);
  h8v c0 = *(const h8v*)(b0);
  h8v c1 = *(const h8v*)(b0 + 16);
  v16h a;
#pragma unroll
  for (int e = 0; e < 8; e++){ a[e] = c0[e]; a[e + 8] = c1[e]; }
  return a;
}
// A fragment from f32 row (global): 4 x b128 + convert.
DEVI v16h a_from_f32p(const float* rowk, int lane){
  const float* b0 = rowk + ((lane >> 4) << 3);
  f4v c0 = *(const f4v*)(b0);
  f4v c1 = *(const f4v*)(b0 + 4);
  f4v c2 = *(const f4v*)(b0 + 16);
  f4v c3 = *(const f4v*)(b0 + 20);
  v16h a;
#pragma unroll
  for (int e = 0; e < 4; e++){
    a[e]      = (_Float16)c0[e];
    a[e + 4]  = (_Float16)c1[e];
    a[e + 8]  = (_Float16)c2[e];
    a[e + 12] = (_Float16)c3[e];
  }
  return a;
}

// LayerNorm(+affine)+ReLU of a 16x128 f32 tile in sH1 -> f16 tile sT. 256 threads.
DEVI void ln_relu_128(const float* sH1, _Float16* sT, const float* g, const float* be,
                      float* sMV, int tid){
  if (tid < 16){
    float s = 0.f, q = 0.f;
    const f4v* row = (const f4v*)(sH1 + tid * 128);
    for (int c = 0; c < 32; c++){
      f4v x = row[c];
      s += x[0] + x[1] + x[2] + x[3];
      q += x[0]*x[0] + x[1]*x[1] + x[2]*x[2] + x[3]*x[3];
    }
    float m = s * (1.0f / 128.0f);
    float v = q * (1.0f / 128.0f) - m * m;
    sMV[tid] = m; sMV[16 + tid] = rsqrtf(v + 1e-5f);
  }
  __syncthreads();
  {
    int i0 = tid * 8;               // 2048 elements / 256 threads = 8 contiguous each
    int r = i0 >> 7, c0 = i0 & 127;
    f4v x0 = *(const f4v*)(sH1 + i0);
    f4v x1 = *(const f4v*)(sH1 + i0 + 4);
    float m = sMV[r], rs = sMV[16 + r];
    h8v o;
#pragma unroll
    for (int j = 0; j < 4; j++){
      o[j]     = (_Float16)fmaxf((x0[j] - m) * rs * g[c0 + j]     + be[c0 + j],     0.f);
      o[j + 4] = (_Float16)fmaxf((x1[j] - m) * rs * g[c0 + 4 + j] + be[c0 + 4 + j], 0.f);
    }
    *(h8v*)(sT + i0) = o;
  }
  __syncthreads();
}

// ---------------- parameter / workspace structs ----------------
struct Par {
  const float *h, *pos, *dp; const int *i0, *i1;
  const float *b1n,*gn,*ben,*b2n;
  const float *b1e,*gee,*bee,*b2e;
  const float *bm;
  const float *b1g,*gg,*beg,*b2g;
  const float *lng,*lnb,*bo;
  const float *b1t,*gt,*bet,*W2t,*b2t;
  const float *b1q,*gq,*beq,*W2q,*b2q;
  const float *W1a,*b1a,*ga,*bea,*W2a,*b2a;
};
struct GemmWs {
  const _Float16 *W1n,*W2n,*W1e,*W2e,*W1g,*W2g,*Wm,*Wo,*W1t,*W1q;
};
struct Bufs {
  float *pos_dom, *trans_dom, *torque_dom, *aggr, *h_dom;
  _Float16 *h_domh;
  int *dom_start;
  float *zf32;          // >= 608 zero floats
  _Float16 *zf16;       // >= 608 zero halfs
};

// ---------------- small kernels ----------------
// Convert f32 weight (srcRows x cols) to f16 packed fragment order, zero-padded to Kpad rows.
__global__ void k_cvt_pack(const float* __restrict__ src, _Float16* __restrict__ dst,
                           int srcRows, int Kpad, int cols){
  int i = blockIdx.x * blockDim.x + threadIdx.x;
  int tot = Kpad * cols;
  if (i >= tot) return;
  int e    = i & 15;
  int lane = (i >> 4) & 31;
  int rest = i >> 9;                 // ks*nct + ct
  int nct  = cols >> 4;
  int ks   = rest / nct, ct = rest - ks * nct;
  int k = ks * 32 + ((lane >> 4) << 3) + (e & 7) + ((e >> 3) << 4);
  int n = ct * 16 + (lane & 15);
  dst[i] = (k < srcRows) ? (_Float16)src[(size_t)k * cols + n] : (_Float16)0.f;
}

// Elementwise f32 -> f16, 8 per thread (vectorized).
__global__ void k_cvt_h(const float* __restrict__ src, _Float16* __restrict__ dst, int n8){
  int i = blockIdx.x * blockDim.x + threadIdx.x;
  if (i >= n8) return;
  const f4v* s = (const f4v*)(src + (size_t)i * 8);
  f4v a = s[0], b = s[1];
  h8v o;
#pragma unroll
  for (int j = 0; j < 4; j++){ o[j] = (_Float16)a[j]; o[j + 4] = (_Float16)b[j]; }
  *(h8v*)(dst + (size_t)i * 8) = o;
}

__global__ void k_zero(float* __restrict__ zf, _Float16* __restrict__ zh, int n){
  int i = blockIdx.x * blockDim.x + threadIdx.x;
  if (i < n){ zf[i] = 0.f; zh[i] = (_Float16)0.f; }
}

__global__ void k_domstart(const int* __restrict__ i0, int* __restrict__ dom_start, int Nn, int Dn){
  int d = blockIdx.x * blockDim.x + threadIdx.x;
  if (d > Dn) return;
  if (d == Dn){ dom_start[Dn] = Nn; return; }
  int lo = 0, hi = Nn;
  while (lo < hi){ int mid = (lo + hi) >> 1; if (i0[mid] < d) lo = mid + 1; else hi = mid; }
  dom_start[d] = lo;
}

__global__ void k_centroid(Par p, Bufs bf, int Dn){
  int gw = (blockIdx.x * blockDim.x + threadIdx.x) >> 5;
  int lane = threadIdx.x & 31;
  if (gw >= Dn) return;
  int s = bf.dom_start[gw], e = bf.dom_start[gw + 1];
  float sx = 0.f, sy = 0.f, sz = 0.f;
  for (int r = s + lane; r < e; r += 32){
    int nd = p.i1[r];
    sx += p.pos[nd * 3 + 0]; sy += p.pos[nd * 3 + 1]; sz += p.pos[nd * 3 + 2];
  }
  for (int o = 16; o > 0; o >>= 1){
    sx += __shfl_down(sx, o, 32); sy += __shfl_down(sy, o, 32); sz += __shfl_down(sz, o, 32);
  }
  if (lane == 0){
    float c = fmaxf((float)(e - s), 1.f);
    bf.pos_dom[gw * 3 + 0] = sx / c; bf.pos_dom[gw * 3 + 1] = sy / c; bf.pos_dom[gw * 3 + 2] = sz / c;
  }
}

// ---------------- fused NodeBlock: hn, he, gate, msg, segment-sum ----------------
template<bool H16>
__global__ __launch_bounds__(256) void k_nodeblock(Par p, GemmWs wg, Bufs bf,
                                                   const _Float16* __restrict__ h16, int Dn){
  const int tid = threadIdx.x, lane = tid & 31, w = tid >> 5;  // wave w -> col-tile w
  const int d0 = blockIdx.x * 2;
  const int s0 = bf.dom_start[d0];
  const int s1 = bf.dom_start[imin(d0 + 1, Dn)];
  const int s2 = bf.dom_start[imin(d0 + 2, Dn)];

  __shared__ alignas(16) _Float16 sE[112 * 64];     // h_edge tiles (f16)
  __shared__ alignas(16) _Float16 sP[112 * 128];    // hn, then P = he*hn
  __shared__ alignas(16) float    sH1[16 * 128];    // pre-LN scratch
  __shared__ alignas(16) _Float16 sT[16 * 128];     // post-LN f16 tile (A of 2nd GEMM)
  __shared__ alignas(16) float    sG[16 * 128];     // sigmoid(gate)
  __shared__ float    sAg[256];                     // per-domain aggr (2 x 128)
  __shared__ float    sMV[32];
  __shared__ float    sDist[112];

  sAg[tid] = 0.f;
  __syncthreads();

  const int n15 = lane & 15, hib = (lane >> 4) << 3;
  const int col = w * 16 + n15;

  for (int B0 = s0; B0 < s2; B0 += 112){
    const int nr = imin(112, s2 - B0);

    // per-tile row base pointers (zero-row for padded rows)
    const float*    baseF[7];
    const _Float16* baseH[7];
#pragma unroll
    for (int t = 0; t < 7; t++){
      int rr = t * 16 + n15;
      bool val = rr < nr;
      int nd = p.i1[val ? (B0 + rr) : B0];
      if (H16) baseH[t] = val ? (h16 + (size_t)nd * 256) : bf.zf16;
      else     baseF[t] = val ? (p.h + (size_t)nd * 256) : bf.zf32;
    }
    auto loadA = [&](int t, int koff) -> v16h {
      if (H16) return a_from_f16p(baseH[t] + koff, lane);
      else     return a_from_f32p(baseF[t] + koff, lane);
    };

    // per-row dist
    for (int r = tid; r < 112; r += 256){
      float dv = 0.f;
      if (r < nr){
        int gr = B0 + r, nd = p.i1[gr];
        int dd = (gr < s1) ? d0 : (d0 + 1);
        float dx = p.pos[nd * 3 + 0] - bf.pos_dom[dd * 3 + 0];
        float dy = p.pos[nd * 3 + 1] - bf.pos_dom[dd * 3 + 1];
        float dz = p.pos[nd * 3 + 2] - bf.pos_dom[dd * 3 + 2];
        dv = sqrtf(dx * dx + dy * dy + dz * dz);
      }
      sDist[r] = dv;
    }
    __syncthreads();
    // Gaussian smearing -> sE (invalid rows get zeros)
    const float stepg = 10.0f / 63.0f;
    const float coeff = -0.5f / (stepg * stepg);
    for (int i = tid; i < 112 * 64; i += 256){
      int r = i >> 6, j = i & 63;
      float v = 0.f;
      if (r < nr){ float dd = sDist[r] - (float)j * stepg; v = expf(coeff * dd * dd); }
      sE[i] = (_Float16)v;
    }
    __syncthreads();

    v8f acc[7];

    // ================= hn = mlp2(h_node) =================
#pragma unroll
    for (int t = 0; t < 7; t++)
#pragma unroll
      for (int r = 0; r < 8; r++) acc[t][r] = 0.f;

    for (int ks = 0; ks < 8; ks++){              // K = 256
      v16h b = load_bp(wg.W1n, 8, ks, w, lane);
#pragma unroll
      for (int t = 0; t < 7; t++)
        acc[t] = wmma16(loadA(t, ks * 32), b, acc[t]);
    }
    {
      const float bias1 = p.b1n[col], bias2 = p.b2n[col];
#pragma unroll
      for (int t = 0; t < 7; t++){
        __syncthreads();
#pragma unroll
        for (int r = 0; r < 8; r++) sH1[(r + hib) * 128 + col] = acc[t][r] + bias1;
        __syncthreads();
        ln_relu_128(sH1, sT, p.gn, p.ben, sMV, tid);
        v8f c2;
#pragma unroll
        for (int r = 0; r < 8; r++) c2[r] = 0.f;
        for (int ks = 0; ks < 4; ks++)
          c2 = wmma16(a_from_f16p(sT + n15 * 128 + ks * 32, lane), load_bp(wg.W2n, 8, ks, w, lane), c2);
#pragma unroll
        for (int r = 0; r < 8; r++)
          sP[(t * 16 + r + hib) * 128 + col] = (_Float16)(c2[r] + bias2);
      }
    }
    __syncthreads();

    // ================= he = mlp2(h_edge); P = he*hn =================
#pragma unroll
    for (int t = 0; t < 7; t++)
#pragma unroll
      for (int r = 0; r < 8; r++) acc[t][r] = 0.f;
    for (int ks = 0; ks < 2; ks++){              // K = 64
      v16h b = load_bp(wg.W1e, 8, ks, w, lane);
#pragma unroll
      for (int t = 0; t < 7; t++)
        acc[t] = wmma16(a_from_f16p(sE + (t * 16 + n15) * 64 + ks * 32, lane), b, acc[t]);
    }
    {
      const float bias1 = p.b1e[col], bias2 = p.b2e[col];
#pragma unroll
      for (int t = 0; t < 7; t++){
        __syncthreads();
#pragma unroll
        for (int r = 0; r < 8; r++) sH1[(r + hib) * 128 + col] = acc[t][r] + bias1;
        __syncthreads();
        ln_relu_128(sH1, sT, p.gee, p.bee, sMV, tid);
        v8f c2;
#pragma unroll
        for (int r = 0; r < 8; r++) c2[r] = 0.f;
        for (int ks = 0; ks < 4; ks++)
          c2 = wmma16(a_from_f16p(sT + n15 * 128 + ks * 32, lane), load_bp(wg.W2e, 8, ks, w, lane), c2);
#pragma unroll
        for (int r = 0; r < 8; r++){
          int ci = (t * 16 + r + hib) * 128 + col;
          sP[ci] = (_Float16)((float)sP[ci] * (c2[r] + bias2));   // same thread wrote hn here
        }
      }
    }
    __syncthreads();

    // ===== gate = mlp2([E | X | 0]); msg = (P@Wm+b)*sigmoid(gate) =====
    // K-regions are k-step aligned: ks 0..1 -> sE, ks 2..9 -> h_node; ks 10 is all-zero, skipped.
#pragma unroll
    for (int t = 0; t < 7; t++)
#pragma unroll
      for (int r = 0; r < 8; r++) acc[t][r] = 0.f;
    for (int ks = 0; ks < 10; ks++){
      v16h b = load_bp(wg.W1g, 8, ks, w, lane);
#pragma unroll
      for (int t = 0; t < 7; t++){
        v16h a = (ks < 2) ? a_from_f16p(sE + (t * 16 + n15) * 64 + ks * 32, lane)
                          : loadA(t, ks * 32 - 64);
        acc[t] = wmma16(a, b, acc[t]);
      }
    }
    {
      const float bias1 = p.b1g[col], bias2 = p.b2g[col], biasm = p.bm[col];
#pragma unroll
      for (int t = 0; t < 7; t++){
        __syncthreads();
#pragma unroll
        for (int r = 0; r < 8; r++) sH1[(r + hib) * 128 + col] = acc[t][r] + bias1;
        __syncthreads();
        ln_relu_128(sH1, sT, p.gg, p.beg, sMV, tid);
        v8f c2;
#pragma unroll
        for (int r = 0; r < 8; r++) c2[r] = 0.f;
        for (int ks = 0; ks < 4; ks++)
          c2 = wmma16(a_from_f16p(sT + n15 * 128 + ks * 32, lane), load_bp(wg.W2g, 8, ks, w, lane), c2);
#pragma unroll
        for (int r = 0; r < 8; r++)
          sG[(r + hib) * 128 + col] = 1.f / (1.f + expf(-(c2[r] + bias2)));
        // msg
        v8f mc;
#pragma unroll
        for (int r = 0; r < 8; r++) mc[r] = 0.f;
        for (int ks = 0; ks < 4; ks++)
          mc = wmma16(a_from_f16p(sP + (t * 16 + n15) * 128 + ks * 32, lane),
                      load_bp(wg.Wm, 8, ks, w, lane), mc);
        __syncthreads();
#pragma unroll
        for (int r = 0; r < 8; r++)
          sH1[(r + hib) * 128 + col] = (mc[r] + biasm) * sG[(r + hib) * 128 + col];
        __syncthreads();
        if (tid < 128){
          for (int r16 = 0; r16 < 16; r16++){
            int rr = t * 16 + r16;
            if (rr < nr){
              int dd = ((B0 + rr) < s1) ? 0 : 1;
              sAg[dd * 128 + tid] += sH1[r16 * 128 + tid];
            }
          }
        }
        __syncthreads();
      }
    }
  } // chunk loop

  __syncthreads();
  {
    int dd = tid >> 7, c = tid & 127, d = d0 + dd;
    if (d < Dn) bf.aggr[(size_t)d * 128 + c] = sAg[tid];
  }
}

// ---------------- h_dom = relu(LN(aggr)) @ Wout + b ----------------
__global__ __launch_bounds__(256) void k_hdom(Par p, GemmWs wg, Bufs bf, int Dn){
  const int tid = threadIdx.x, lane = tid & 31, w = tid >> 5;
  const int r0 = blockIdx.x * 16;
  __shared__ alignas(16) float    sH1[16 * 128];
  __shared__ alignas(16) _Float16 sT[16 * 128];
  __shared__ float    sMV[32];
  for (int i = tid; i < 2048; i += 256){
    int r = i >> 7, c = i & 127;
    int gr = r0 + r;
    sH1[i] = (gr < Dn) ? bf.aggr[(size_t)gr * 128 + c] : 0.f;
  }
  __syncthreads();
  ln_relu_128(sH1, sT, p.lng, p.lnb, sMV, tid);
  const int n15 = lane & 15, hib = (lane >> 4) << 3;
  for (int ct = w * 2; ct < w * 2 + 2; ct++){
    v8f c2;
#pragma unroll
    for (int r = 0; r < 8; r++) c2[r] = 0.f;
    for (int ks = 0; ks < 4; ks++)
      c2 = wmma16(a_from_f16p(sT + n15 * 128 + ks * 32, lane), load_bp(wg.Wo, 16, ks, ct, lane), c2);
#pragma unroll
    for (int r = 0; r < 8; r++){
      int gr = r0 + r + hib, col = ct * 16 + n15;
      if (gr < Dn){
        float v = c2[r] + p.bo[col];
        bf.h_dom[(size_t)gr * 256 + col]  = v;
        bf.h_domh[(size_t)gr * 256 + col] = (_Float16)v;
      }
    }
  }
}

// ---------------- trans + torq scalars and per-domain means ----------------
template<bool H16>
__global__ __launch_bounds__(256) void k_transtorq(Par p, GemmWs wg, Bufs bf,
                                                   const _Float16* __restrict__ h16, int Dn){
  const int tid = threadIdx.x, lane = tid & 31, w = tid >> 5;
  const int d0 = blockIdx.x * 2;
  const int s0 = bf.dom_start[d0];
  const int s1 = bf.dom_start[imin(d0 + 1, Dn)];
  const int s2 = bf.dom_start[imin(d0 + 2, Dn)];

  __shared__ alignas(16) _Float16 sE[112 * 64];
  __shared__ float    sGeo[112 * 12]; // radius(3) dp(3) torque(3) dist ndp ntq
  __shared__ alignas(16) float    sH1[16 * 128];
  __shared__ alignas(16) _Float16 sT[16 * 128];
  __shared__ float    sMV[32];
  __shared__ float    sTW[112];
  __shared__ float    sSC[112];
  __shared__ float    sSum[12];

  if (tid < 12) sSum[tid] = 0.f;
  __syncthreads();

  const int n15 = lane & 15, hib = (lane >> 4) << 3;
  const int col = w * 16 + n15;

  for (int B0 = s0; B0 < s2; B0 += 112){
    const int nr = imin(112, s2 - B0);

    // per-tile row base pointers
    const float*    baseF[7];
    const _Float16* baseH[7];
    const _Float16* baseHd[7];
#pragma unroll
    for (int t = 0; t < 7; t++){
      int rr = t * 16 + n15;
      bool val = rr < nr;
      int gr = B0 + rr;
      int nd = p.i1[val ? gr : B0];
      int dd = (val && gr >= s1) ? (d0 + 1) : d0;
      if (H16) baseH[t] = val ? (h16 + (size_t)nd * 256) : bf.zf16;
      else     baseF[t] = val ? (p.h + (size_t)nd * 256) : bf.zf32;
      baseHd[t] = val ? (bf.h_domh + (size_t)dd * 256) : bf.zf16;
    }
    auto loadA = [&](int t, int koff) -> v16h {
      if (H16) return a_from_f16p(baseH[t] + koff, lane);
      else     return a_from_f32p(baseF[t] + koff, lane);
    };

    for (int r = tid; r < 112; r += 256){
      float rad[3] = {0,0,0}, dpv[3] = {0,0,0}, tq[3] = {0,0,0}, dv = 0, ndp = 0, ntq = 0;
      if (r < nr){
        int gr = B0 + r, nd = p.i1[gr], dd = (gr < s1) ? d0 : (d0 + 1);
        for (int c = 0; c < 3; c++){
          rad[c] = p.pos[nd * 3 + c] - bf.pos_dom[dd * 3 + c];
          dpv[c] = p.dp[nd * 3 + c];
        }
        tq[0] = rad[1] * dpv[2] - rad[2] * dpv[1];
        tq[1] = rad[2] * dpv[0] - rad[0] * dpv[2];
        tq[2] = rad[0] * dpv[1] - rad[1] * dpv[0];
        dv  = sqrtf(rad[0]*rad[0] + rad[1]*rad[1] + rad[2]*rad[2]);
        ndp = sqrtf(dpv[0]*dpv[0] + dpv[1]*dpv[1] + dpv[2]*dpv[2]);
        ntq = sqrtf(tq[0]*tq[0] + tq[1]*tq[1] + tq[2]*tq[2]);
      }
      float* g = &sGeo[r * 12];
      g[0]=rad[0]; g[1]=rad[1]; g[2]=rad[2];
      g[3]=dpv[0]; g[4]=dpv[1]; g[5]=dpv[2];
      g[6]=tq[0];  g[7]=tq[1];  g[8]=tq[2];
      g[9]=dv; g[10]=ndp; g[11]=ntq;
    }
    __syncthreads();
    const float stepg = 10.0f / 63.0f;
    const float coeff = -0.5f / (stepg * stepg);
    for (int i = tid; i < 112 * 64; i += 256){
      int r = i >> 6, j = i & 63;
      float v = 0.f;
      if (r < nr){ float dd = sGeo[r * 12 + 9] - (float)j * stepg; v = expf(coeff * dd * dd); }
      sE[i] = (_Float16)v;
    }
    __syncthreads();

    v8f acc[7];

    // ======== trans: K = 576 ([h_dom | h_in | h_edge]) ========
#pragma unroll
    for (int t = 0; t < 7; t++)
#pragma unroll
      for (int r = 0; r < 8; r++) acc[t][r] = 0.f;
    for (int ks = 0; ks < 18; ks++){
      v16h b = load_bp(wg.W1t, 8, ks, w, lane);
#pragma unroll
      for (int t = 0; t < 7; t++){
        v16h a;
        if (ks < 8)       a = a_from_f16p(baseHd[t] + ks * 32, lane);
        else if (ks < 16) a = loadA(t, ks * 32 - 256);
        else              a = a_from_f16p(sE + (t * 16 + n15) * 64 + (ks - 16) * 32, lane);
        acc[t] = wmma16(a, b, acc[t]);
      }
    }
    {
      const float bias1 = p.b1t[col];
#pragma unroll
      for (int t = 0; t < 7; t++){
        __syncthreads();
#pragma unroll
        for (int r = 0; r < 8; r++) sH1[(r + hib) * 128 + col] = acc[t][r] + bias1;
        __syncthreads();
        ln_relu_128(sH1, sT, p.gt, p.bet, sMV, tid);
        for (int rr2 = 0; rr2 < 2; rr2++){
          int r = w * 2 + rr2;
          float pp = 0.f;
          for (int j = lane; j < 128; j += 32) pp += (float)sT[r * 128 + j] * p.W2t[j];
          for (int o = 16; o > 0; o >>= 1) pp += __shfl_down(pp, o, 32);
          if (lane == 0) sTW[t * 16 + r] = pp + p.b2t[0];
        }
      }
    }
    __syncthreads();

    // ======== torq: K = 544 (515 padded: [h_in | h_dom | dist | ndp | ntq]) ========
#pragma unroll
    for (int t = 0; t < 7; t++)
#pragma unroll
      for (int r = 0; r < 8; r++) acc[t][r] = 0.f;
    for (int ks = 0; ks < 17; ks++){
      v16h b = load_bp(wg.W1q, 8, ks, w, lane);
#pragma unroll
      for (int t = 0; t < 7; t++){
        int rr = t * 16 + n15;
        v16h a;
        if (ks < 8)       a = loadA(t, ks * 32);
        else if (ks < 16) a = a_from_f16p(baseHd[t] + ks * 32 - 256, lane);
        else {
          // k = 512 + akk(e,lane); only k=512,513,514 are nonzero (dist, |dp|, |torque|)
          bool val = rr < nr;
#pragma unroll
          for (int e = 0; e < 16; e++){
            int k = akk(e, lane);
            _Float16 xv = (_Float16)0.f;
            if (val && k < 3) xv = (_Float16)sGeo[rr * 12 + 9 + k];
            a[e] = xv;
          }
        }
        acc[t] = wmma16(a, b, acc[t]);
      }
    }
    {
      const float bias1 = p.b1q[col];
#pragma unroll
      for (int t = 0; t < 7; t++){
        __syncthreads();
#pragma unroll
        for (int r = 0; r < 8; r++) sH1[(r + hib) * 128 + col] = acc[t][r] + bias1;
        __syncthreads();
        ln_relu_128(sH1, sT, p.gq, p.beq, sMV, tid);
        for (int rr2 = 0; rr2 < 2; rr2++){
          int r = w * 2 + rr2;
          float pp = 0.f;
          for (int j = lane; j < 128; j += 32) pp += (float)sT[r * 128 + j] * p.W2q[j];
          for (int o = 16; o > 0; o >>= 1) pp += __shfl_down(pp, o, 32);
          if (lane == 0) sSC[t * 16 + r] = pp + p.b2q[0];
        }
      }
    }
    __syncthreads();

    // ======== deterministic per-domain sums ========
    if (tid < 12){
      int dd = tid / 6, c = tid % 6;
      float s = 0.f;
      for (int r = 0; r < nr; r++){
        int rd = ((B0 + r) < s1) ? 0 : 1;
        if (rd != dd) continue;
        if (c < 3) s += sTW[r] * sGeo[r * 12 + 3 + c];
        else       s += sSC[r] * sGeo[r * 12 + 6 + (c - 3)];
      }
      sSum[tid] += s;
    }
    __syncthreads();
  }

  if (tid < 12){
    int dd = tid / 6, c = tid % 6, d = d0 + dd;
    if (d < Dn){
      float cnt = fmaxf((float)(bf.dom_start[d + 1] - bf.dom_start[d]), 1.f);
      float v = sSum[tid] / cnt;
      if (c < 3) bf.trans_dom[d * 3 + c]     = v;
      else       bf.torque_dom[d * 3 + (c-3)] = v;
    }
  }
}

// ---------------- ang MLP (f32) + rot axis/angle outputs ----------------
__global__ void k_ang(Par p, Bufs bf, float* __restrict__ out_axis, float* __restrict__ out_ang, int Dn){
  int gw = (blockIdx.x * blockDim.x + threadIdx.x) >> 5;
  int lane = threadIdx.x & 31;
  if (gw >= Dn) return;
  float tx = bf.torque_dom[gw * 3 + 0], ty = bf.torque_dom[gw * 3 + 1], tz = bf.torque_dom[gw * 3 + 2];
  float tn = sqrtf(tx * tx + ty * ty + tz * tz);
  if (lane == 0){
    out_axis[gw * 3 + 0] = tx / tn; out_axis[gw * 3 + 1] = ty / tn; out_axis[gw * 3 + 2] = tz / tn;
  }
  const float* hd = bf.h_dom + (size_t)gw * 256;
  float h0 = p.b1a[lane], h1 = p.b1a[lane + 32];
  for (int k = 0; k < 257; k++){
    float xv = (k < 256) ? hd[k] : tn;
    h0 += xv * p.W1a[k * 64 + lane];
    h1 += xv * p.W1a[k * 64 + lane + 32];
  }
  float s = h0 + h1, q = h0 * h0 + h1 * h1;
  for (int o = 16; o > 0; o >>= 1){ s += __shfl_down(s, o, 32); q += __shfl_down(q, o, 32); }
  s = __shfl(s, 0, 32); q = __shfl(q, 0, 32);
  float m = s * (1.f / 64.f), vv = q * (1.f / 64.f) - m * m, rs = rsqrtf(vv + 1e-5f);
  float t0 = fmaxf((h0 - m) * rs * p.ga[lane]      + p.bea[lane],      0.f);
  float t1 = fmaxf((h1 - m) * rs * p.ga[lane + 32] + p.bea[lane + 32], 0.f);
  float pp = t0 * p.W2a[lane] + t1 * p.W2a[lane + 32];
  for (int o = 16; o > 0; o >>= 1) pp += __shfl_down(pp, o, 32);
  if (lane == 0)
    out_ang[gw] = (1.f / (1.f + expf(-(pp + p.b2a[0])))) * 3.14159265358979323846f;
}

__global__ void k_base(const float* __restrict__ pos, const float* __restrict__ dpv,
                       float* __restrict__ outp, int tot){
  int i = blockIdx.x * blockDim.x + threadIdx.x;
  if (i >= tot) return;
  outp[i] = pos[i] + dpv[i];
}

__global__ void k_final(Par p, Bufs bf, const float* __restrict__ out_axis,
                        const float* __restrict__ out_ang, float* __restrict__ outp, int Nn){
  int r = blockIdx.x * blockDim.x + threadIdx.x;
  if (r >= Nn) return;
  int nd = p.i1[r], d = p.i0[r];
  float rx = p.pos[nd * 3 + 0] - bf.pos_dom[d * 3 + 0];
  float ry = p.pos[nd * 3 + 1] - bf.pos_dom[d * 3 + 1];
  float rz = p.pos[nd * 3 + 2] - bf.pos_dom[d * 3 + 2];
  float ax = out_axis[d * 3 + 0], ay = out_axis[d * 3 + 1], az = out_axis[d * 3 + 2];
  float an = out_ang[d];
  float c = cosf(an), s = sinf(an);
  float cx = ay * rz - az * ry, cy = az * rx - ax * rz, cz = ax * ry - ay * rx;
  float k = (1.f - c) * (ax * rx + ay * ry + az * rz);
  outp[nd * 3 + 0] = bf.pos_dom[d * 3 + 0] + bf.trans_dom[d * 3 + 0] + rx * c + cx * s + ax * k;
  outp[nd * 3 + 1] = bf.pos_dom[d * 3 + 1] + bf.trans_dom[d * 3 + 1] + ry * c + cy * s + ay * k;
  outp[nd * 3 + 2] = bf.pos_dom[d * 3 + 2] + bf.trans_dom[d * 3 + 2] + rz * c + cz * s + az * k;
}

// ---------------- host: map inputs, lay out workspace, launch pipeline ----------------
extern "C" void kernel_launch(void* const* d_in, const int* in_sizes, int n_in,
                              void* d_out, int out_size, void* d_ws, size_t ws_size,
                              hipStream_t stream){
  (void)n_in;
  auto F  = [&](int i)->const float*{ return (const float*)d_in[i]; };
  auto Ii = [&](int i)->const int*  { return (const int*)d_in[i]; };

  Par P{};
  const float *W1n, *W2n, *W1e, *W2e, *W1g, *W2g, *Wm, *Wo, *W1t, *W1q;
  bool layB = (in_sizes[0] < in_sizes[3]);   // full-tree-sorted flatten
  int pb;
  if (layB){ P.dp = F(0); P.i0 = Ii(1); P.i1 = Ii(2); P.h = F(3); P.pos = F(46); pb = 4; }
  else     { P.h = F(0); P.pos = F(1); P.dp = F(2); P.i0 = Ii(3); P.i1 = Ii(4); pb = 5; }
  bool sortedP = layB || (in_sizes[5] == 16448 /* ang.W1 = 257*64 */);
  if (sortedP){
    int b = pb; // params sorted: ang, nb_edge, nb_gate, nb_ln_b, nb_ln_g, nb_msg_W, nb_msg_b,
                //                nb_node, nb_out_W, nb_out_b, torq, trans; mlp: W1,W2,b1,b2,be,g
    P.W1a=F(b+0);  P.W2a=F(b+1);  P.b1a=F(b+2);  P.b2a=F(b+3);  P.bea=F(b+4);  P.ga=F(b+5);
    W1e=F(b+6);    W2e=F(b+7);    P.b1e=F(b+8);  P.b2e=F(b+9);  P.bee=F(b+10); P.gee=F(b+11);
    W1g=F(b+12);   W2g=F(b+13);   P.b1g=F(b+14); P.b2g=F(b+15); P.beg=F(b+16); P.gg=F(b+17);
    P.lnb=F(b+18); P.lng=F(b+19);
    Wm=F(b+20);    P.bm=F(b+21);
    W1n=F(b+22);   W2n=F(b+23);   P.b1n=F(b+24); P.b2n=F(b+25); P.ben=F(b+26); P.gn=F(b+27);
    Wo=F(b+28);    P.bo=F(b+29);
    W1q=F(b+30);   P.W2q=F(b+31); P.b1q=F(b+32); P.b2q=F(b+33); P.beq=F(b+34); P.gq=F(b+35);
    W1t=F(b+36);   P.W2t=F(b+37); P.b1t=F(b+38); P.b2t=F(b+39); P.bet=F(b+40); P.gt=F(b+41);
  } else {        // recursive insertion order; mlp: W1,b1,g,be,W2,b2
    W1n=F(5);  P.b1n=F(6);  P.gn=F(7);   P.ben=F(8);  W2n=F(9);  P.b2n=F(10);
    W1e=F(11); P.b1e=F(12); P.gee=F(13); P.bee=F(14); W2e=F(15); P.b2e=F(16);
    Wm=F(17);  P.bm=F(18);
    W1g=F(19); P.b1g=F(20); P.gg=F(21);  P.beg=F(22); W2g=F(23); P.b2g=F(24);
    P.lng=F(25); P.lnb=F(26);
    Wo=F(27);  P.bo=F(28);
    W1t=F(29); P.b1t=F(30); P.gt=F(31);  P.bet=F(32); P.W2t=F(33); P.b2t=F(34);
    W1q=F(35); P.b1q=F(36); P.gq=F(37);  P.beq=F(38); P.W2q=F(39); P.b2q=F(40);
    P.W1a=F(41); P.b1a=F(42); P.ga=F(43); P.bea=F(44); P.W2a=F(45); P.b2a=F(46);
  }

  const int Nn = (layB ? in_sizes[0] : in_sizes[1]) / 3;   // pos_node has N*3 elems
  const int Dn = (out_size - 3 * Nn) / 4;                  // out = 3N + 3D + D

  // workspace layout
  char* ws = (char*)d_ws;
  size_t off = 0;
  auto alloc = [&](size_t bytes)->void*{
    void* q = (void*)(ws + off);
    off = (off + bytes + 255) & ~(size_t)255;
    return q;
  };
  _Float16* w1n = (_Float16*)alloc((size_t)32768 * 2);
  _Float16* w2n = (_Float16*)alloc((size_t)16384 * 2);
  _Float16* w1e = (_Float16*)alloc((size_t)8192  * 2);
  _Float16* w2e = (_Float16*)alloc((size_t)16384 * 2);
  _Float16* w1g = (_Float16*)alloc((size_t)45056 * 2);  // 352x128 packed, rows>=321 zero
  _Float16* w2g = (_Float16*)alloc((size_t)16384 * 2);
  _Float16* wm  = (_Float16*)alloc((size_t)16384 * 2);
  _Float16* wo  = (_Float16*)alloc((size_t)32768 * 2);
  _Float16* w1t = (_Float16*)alloc((size_t)73728 * 2);
  _Float16* w1q = (_Float16*)alloc((size_t)69632 * 2);  // 544x128 packed, rows>=515 zero
  Bufs B{};
  B.pos_dom    = (float*)alloc((size_t)Dn * 3 * 4);
  B.trans_dom  = (float*)alloc((size_t)Dn * 3 * 4);
  B.torque_dom = (float*)alloc((size_t)Dn * 3 * 4);
  B.aggr       = (float*)alloc((size_t)Dn * 128 * 4);
  B.h_dom      = (float*)alloc((size_t)Dn * 256 * 4);
  B.h_domh     = (_Float16*)alloc((size_t)Dn * 256 * 2);
  B.dom_start  = (int*)alloc((size_t)(Dn + 1) * 4);
  B.zf32       = (float*)alloc((size_t)608 * 4);
  B.zf16       = (_Float16*)alloc((size_t)608 * 2);
  // optional f16 copy of h_node (used when workspace is large enough)
  _Float16* h16 = (_Float16*)alloc((size_t)Nn * 256 * 2);
  const bool useH16 = (off <= ws_size);
  GemmWs G{w1n, w2n, w1e, w2e, w1g, w2g, wm, wo, w1t, w1q};

  auto cvt = [&](const float* src, _Float16* dst, int sr, int kpad, int cols){
    int tot = kpad * cols;
    k_cvt_pack<<<(tot + 255) / 256, 256, 0, stream>>>(src, dst, sr, kpad, cols);
  };
  cvt(W1n, w1n, 256, 256, 128);
  cvt(W2n, w2n, 128, 128, 128);
  cvt(W1e, w1e,  64,  64, 128);
  cvt(W2e, w2e, 128, 128, 128);
  cvt(W1g, w1g, 321, 352, 128);
  cvt(W2g, w2g, 128, 128, 128);
  cvt(Wm,  wm,  128, 128, 128);
  cvt(Wo,  wo,  128, 128, 256);
  cvt(W1t, w1t, 576, 576, 128);
  cvt(W1q, w1q, 515, 544, 128);
  k_zero<<<3, 256, 0, stream>>>(B.zf32, B.zf16, 608);
  if (useH16){
    int n8 = Nn * 32;   // Nn*256/8
    k_cvt_h<<<(n8 + 255) / 256, 256, 0, stream>>>(P.h, h16, n8);
  }

  k_domstart<<<(Dn + 1 + 255) / 256, 256, 0, stream>>>(P.i0, B.dom_start, Nn, Dn);
  k_centroid<<<(Dn + 7) / 8, 256, 0, stream>>>(P, B, Dn);
  if (useH16){
    k_nodeblock<true><<<(Dn + 1) / 2, 256, 0, stream>>>(P, G, B, h16, Dn);
  } else {
    k_nodeblock<false><<<(Dn + 1) / 2, 256, 0, stream>>>(P, G, B, nullptr, Dn);
  }
  k_hdom<<<(Dn + 15) / 16, 256, 0, stream>>>(P, G, B, Dn);
  if (useH16){
    k_transtorq<true><<<(Dn + 1) / 2, 256, 0, stream>>>(P, G, B, h16, Dn);
  } else {
    k_transtorq<false><<<(Dn + 1) / 2, 256, 0, stream>>>(P, G, B, nullptr, Dn);
  }

  float* out      = (float*)d_out;
  float* out_axis = out + (size_t)Nn * 3;
  float* out_ang  = out_axis + (size_t)Dn * 3;
  k_ang<<<(Dn + 7) / 8, 256, 0, stream>>>(P, B, out_axis, out_ang, Dn);
  k_base<<<((Nn * 3) + 255) / 256, 256, 0, stream>>>(P.pos, P.dp, out, Nn * 3);
  k_final<<<(Nn + 255) / 256, 256, 0, stream>>>(P, B, out_axis, out_ang, out, Nn);
}
